// GainEncoder_2843268350277
// MI455X (gfx1250) — matrix operand order repaired and checked by reading, not compiled, not run on hardware
//
#include <hip/hip_runtime.h>

#define NUM_GRAPHS 4096
#define NPG 64      // nodes per graph
#define EPG 256     // edges per graph
#define HID 64
#define GB  16      // graphs per block in head kernel

// ---- workspace layout (units of 4 bytes; all offsets multiples of 4) ----
#define HP_OFF   0
#define HP_SZ    (NUM_GRAPHS * 320)          // pooled+terminal feats, f32
#define GW1_OFF  (HP_OFF + HP_SZ)            // packed bf16-pair weight images
#define GA1_OFF  (GW1_OFF + 2048)
#define GW2_OFF  (GA1_OFF + 2048)
#define GA2_OFF  (GW2_OFF + 2048)
#define T1P_OFF  (GA2_OFF + 2048)
#define T2P_OFF  (T1P_OFF + 4096)
#define S1P_OFF  (T2P_OFF + 1024)
#define P1P_OFF  (S1P_OFF + 3072)
#define P2P_OFF  (P1P_OFF + 40960)
#define P3P_OFF  (P2P_OFF + 16384)

typedef __attribute__((ext_vector_type(16))) __bf16       v16bf;
typedef __attribute__((ext_vector_type(8)))  float        v8f;
typedef __attribute__((ext_vector_type(8)))  unsigned int v8u;

__device__ __forceinline__ unsigned int pkbf(float a, float b) {
    unsigned int ua = __builtin_bit_cast(unsigned int, a);
    unsigned int ub = __builtin_bit_cast(unsigned int, b);
    return (ua >> 16) | (ub & 0xFFFF0000u);
}

// A fragment (16-bit, 16x32): VGPR v<4 : K = half*8 + 2v ; v>=4 : +16.
// Packed [row][K/2] layout -> dwords {half*4 .. +3} and {8+half*4 .. +3}.
__device__ __forceinline__ v16bf load_frag_a(const unsigned* __restrict__ row, int hh) {
    const uint4* p = (const uint4*)row;
    const uint4 a0 = p[hh];
    const uint4 a1 = p[hh + 2];
    v8u u; u[0]=a0.x; u[1]=a0.y; u[2]=a0.z; u[3]=a0.w;
           u[4]=a1.x; u[5]=a1.y; u[6]=a1.z; u[7]=a1.w;
    return __builtin_bit_cast(v16bf, u);
}

// B fragment (16-bit, 32x16): VGPR v<4 : K = half*8 + 2v ; v>=4 : +16.
// Packed [n][K/2] layout -> dwords {half*4 .. +3} and {8+half*4 .. +3}.
__device__ __forceinline__ v16bf load_frag_b(const unsigned* __restrict__ row, int hh) {
    const uint4* p = (const uint4*)row;
    const uint4 b0 = p[hh];
    const uint4 b1 = p[hh + 2];
    v8u u; u[0]=b0.x; u[1]=b0.y; u[2]=b0.z; u[3]=b0.w;
           u[4]=b1.x; u[5]=b1.y; u[6]=b1.z; u[7]=b1.w;
    return __builtin_bit_cast(v16bf, u);
}

struct GnnP {
    const float* W0; const float* A0;        // layer-0 f32 weights (K=4, VALU)
    const float* E[3]; const float* b[3];
    const unsigned* Wp[2]; const unsigned* Ap[2];  // packed layer-1/2 weights
};

struct HeadP {
    const unsigned *t1p,*t2p,*s1p,*p1p,*p2p,*p3p;  // packed weight images
    const float *t1b,*t2b,*s1b,*p1b,*p2b,*p3b;
    const float *tmuw,*tmub,*tlvw,*tlvb,*smuw,*smub,*slvw,*slvb;
    const float *pmuw,*pmub,*plvw,*plvb;
    const float *g1w,*g1b,*g2w,*g2b,*g3w,*g3b;
};

// ---------------------------------------------------------------------------
// Weight pre-pack: W[K][N] f32 row-major -> Wp[N][K/2] bf16-pair dwords.
// ---------------------------------------------------------------------------
__global__ __launch_bounds__(256) void pack_kernel(
    const float* __restrict__ src, unsigned* __restrict__ dst, int K, int N)
{
    const int kd = K >> 1;
    const int total = N * kd;
    for (int i = blockIdx.x * 256 + threadIdx.x; i < total; i += gridDim.x * 256) {
        const int n = i / kd, j = i - n * kd;
        dst[i] = pkbf(src[(2 * j) * N + n], src[(2 * j + 1) * N + n]);
    }
}

// ---------------------------------------------------------------------------
// Kernel 1: one block per graph; fused 3-layer edge-conditioned GNN in LDS.
// ---------------------------------------------------------------------------
__global__ __launch_bounds__(256) void gnn_kernel(
    const float* __restrict__ x, const int* __restrict__ ei,
    const float* __restrict__ ea, GnnP P, float* __restrict__ hp)
{
    __shared__ __align__(16) unsigned hbp[NPG * 32];   // h, packed bf16 pairs (8KB)
    __shared__ __align__(16) unsigned wpk[HID * 32];   // packed W or A (8KB)
    __shared__ float hf[NPG * HID];                    // final h, f32 (16KB)
    __shared__ float agg[NPG * HID];                   // accumulator (16KB)
    __shared__ float el[4 * HID];                      // E (f32)
    __shared__ float w0[4 * HID];                      // layer-0 W/A (f32)
    __shared__ float xl[NPG * 4];
    __shared__ float eat[EPG * 4];
    __shared__ int   srcl[EPG], dstl[EPG];
    __shared__ int   lastt[4];

    const int g    = blockIdx.x;
    const int tid  = threadIdx.x;
    const int lane = tid & 31;
    const int wave = tid >> 5;
    const int lr   = lane & 15;
    const int hh   = lane >> 4;

    // ---- stage per-graph inputs ----
    xl[tid]   = x[g * NPG * 4 + tid];
    srcl[tid] = ei[g * EPG + tid] & (NPG - 1);
    dstl[tid] = ei[(size_t)NUM_GRAPHS * EPG + g * EPG + tid] & (NPG - 1);
    for (int i = tid; i < EPG * 4; i += 256) eat[i] = ea[g * EPG * 4 + i];
    __syncthreads();

    // =================== layer 0 (K = 4, VALU) ===================
    for (int i = tid; i < 4 * HID; i += 256) { w0[i] = P.W0[i]; el[i] = P.E[0][i]; }
    __syncthreads();
    for (int i = tid; i < NPG * HID; i += 256) {
        const int m = i >> 6, f = i & 63;
        const float* xr = &xl[m * 4];
        agg[i] = xr[0]*w0[f] + xr[1]*w0[64+f] + xr[2]*w0[128+f] + xr[3]*w0[192+f];
    }
    __syncthreads();
    for (int i = tid; i < 4 * HID; i += 256) w0[i] = P.A0[i];
    __syncthreads();
    {   // one edge per thread
        const int e = tid;
        const int s = srcl[e], d = dstl[e];
        const float x0 = xl[s*4], x1 = xl[s*4+1], x2 = xl[s*4+2], x3 = xl[s*4+3];
        const float e0 = eat[e*4], e1 = eat[e*4+1], e2 = eat[e*4+2], e3 = eat[e*4+3];
        for (int f = 0; f < HID; ++f) {
            const float v = x0*w0[f] + x1*w0[64+f] + x2*w0[128+f] + x3*w0[192+f]
                          + e0*el[f] + e1*el[64+f] + e2*el[128+f] + e3*el[192+f];
            atomicAdd(&agg[d * HID + f], v);       // ds_add_f32
        }
    }
    __syncthreads();
    // finalize: relu(agg + b) -> packed bf16 h
    for (int i = tid; i < NPG * 32; i += 256) {
        const int m = i >> 5, j = (i & 31) << 1;
        const float v0 = fmaxf(agg[m*HID + j]     + P.b[0][j],     0.f);
        const float v1 = fmaxf(agg[m*HID + j + 1] + P.b[0][j + 1], 0.f);
        hbp[i] = pkbf(v0, v1);
    }
    __syncthreads();

    // =================== layers 1..2 (K = 64, bf16 WMMA) ===================
    for (int L = 1; L < 3; ++L) {
        for (int i = tid; i < HID * 32; i += 256) wpk[i] = P.Wp[L - 1][i];
        for (int i = tid; i < 4 * HID; i += 256)  el[i]  = P.E[L][i];
        __syncthreads();

        // node self-transform: T = h @ W (16 tiles, plain stores into agg)
        for (int t = wave; t < 16; t += 8) {
            const int mr = (t & 3) << 4, nc = (t >> 2) << 4;
            v8f acc = {};
#pragma unroll
            for (int kb = 0; kb < HID; kb += 32) {
                const int kd = kb >> 1;
                const v16bf af = load_frag_a(&hbp[(mr + lr) * 32 + kd], hh);
                const v16bf bf = load_frag_b(&wpk[(nc + lr) * 32 + kd], hh);
                acc = __builtin_amdgcn_wmma_f32_16x16x32_bf16(
                    false, af, false, bf, (short)0, acc, false, false);
            }
            const int n = nc + lr;
#pragma unroll
            for (int i = 0; i < 8; ++i)
                agg[(mr + i + (hh << 3)) * HID + n] = acc[i];
        }
        __syncthreads();

        // edge messages: gather-GEMM h[src] @ A, LDS-atomic scatter by dst
        for (int i = tid; i < HID * 32; i += 256) wpk[i] = P.Ap[L - 1][i];
        __syncthreads();
        for (int t = wave; t < 64; t += 8) {       // 16 edge tiles x 4 feat tiles
            const int er = (t & 15) << 4, nc = (t >> 4) << 4;
            const int srow = srcl[er + lr];        // per-lane row gather
            v8f acc = {};
#pragma unroll
            for (int kb = 0; kb < HID; kb += 32) {
                const int kd = kb >> 1;
                const v16bf af = load_frag_a(&hbp[srow * 32 + kd], hh);
                const v16bf bf = load_frag_b(&wpk[(nc + lr) * 32 + kd], hh);
                acc = __builtin_amdgcn_wmma_f32_16x16x32_bf16(
                    false, af, false, bf, (short)0, acc, false, false);
            }
            const int n = nc + lr;
            const float E0 = el[n], E1 = el[64 + n], E2 = el[128 + n], E3 = el[192 + n];
#pragma unroll
            for (int i = 0; i < 8; ++i) {
                const int e = er + i + (hh << 3);
                const float v = acc[i] + eat[e*4]*E0 + eat[e*4+1]*E1
                                       + eat[e*4+2]*E2 + eat[e*4+3]*E3;
                atomicAdd(&agg[dstl[e] * HID + n], v);   // ds_add_f32 scatter
            }
        }
        __syncthreads();
        // finalize: relu(agg + b) -> packed bf16 (and f32 copy on last layer)
        for (int i = tid; i < NPG * 32; i += 256) {
            const int m = i >> 5, j = (i & 31) << 1;
            const float v0 = fmaxf(agg[m*HID + j]     + P.b[L][j],     0.f);
            const float v1 = fmaxf(agg[m*HID + j + 1] + P.b[L][j + 1], 0.f);
            hbp[i] = pkbf(v0, v1);
            if (L == 2) { hf[m*HID + j] = v0; hf[m*HID + j + 1] = v1; }
        }
        __syncthreads();
    }

    // =================== pooling: mean + max over 64 nodes ===================
    if (tid < HID) {
        float s = 0.f, mx = -3.402823466e38f;
        for (int m = 0; m < NPG; ++m) {
            const float v = hf[m * HID + tid];
            s += v; mx = fmaxf(mx, v);
        }
        hp[(size_t)g * 320 + tid]      = s * (1.f / NPG);
        hp[(size_t)g * 320 + 64 + tid] = mx;
    }
    if (tid < 4) lastt[tid] = -1;
    __syncthreads();
    if (tid < NPG) {   // terminal extraction: last node per type (first-max argmax)
        const float* xr = &xl[tid * 4];
        int nt = 0; float bv = xr[0];
        for (int k = 1; k < 4; ++k) if (xr[k] > bv) { bv = xr[k]; nt = k; }
        if (nt < 3) atomicMax(&lastt[nt], tid);
    }
    __syncthreads();
    if (tid < 192) {
        const int tt = tid / 64, f = tid & 63;
        const int li = lastt[tt];
        hp[(size_t)g * 320 + 128 + tid] = (li >= 0) ? hf[li * HID + f] : 0.f;
    }
}

// ---------------------------------------------------------------------------
// Head GEMM: C = relu(A @ B + bias). A packed bf16 in LDS, B packed in global
// (L2-resident). Output either f32 (feeds VALU heads) or packed bf16 via
// shfl_xor neighbor-column exchange (feeds next GEMM).
// ---------------------------------------------------------------------------
__device__ __forceinline__ void wmma_gemm_pk(
    const unsigned* __restrict__ Ap, int lda,        // dwords
    const unsigned* __restrict__ Bp, int ldb,        // dwords (= K/2)
    const float* __restrict__ bias,
    float* __restrict__ Cf, unsigned* __restrict__ Cp, int ldc,  // cols
    int Mt, int Nt, int K)
{
    const int lane = threadIdx.x & 31;
    const int wave = threadIdx.x >> 5;
    const int lr   = lane & 15;
    const int hh   = lane >> 4;
    const int ntiles = Mt * Nt;
    for (int t = wave; t < ntiles; t += 8) {
        const int mr = (t % Mt) << 4;
        const int nc = (t / Mt) << 4;
        v8f acc = {};
        const unsigned* arow = Ap + (mr + lr) * lda;
        const unsigned* brow = Bp + (nc + lr) * ldb;
        for (int kd = 0; kd < (K >> 1); kd += 16) {
            const v16bf af = load_frag_a(arow + kd, hh);
            const v16bf bf = load_frag_b(brow + kd, hh);
            acc = __builtin_amdgcn_wmma_f32_16x16x32_bf16(
                false, af, false, bf, (short)0, acc, false, false);
        }
        const int n = nc + lr;
        const float bb = bias[n];
#pragma unroll
        for (int i = 0; i < 8; ++i) {
            const int m = mr + i + (hh << 3);
            const float v = fmaxf(acc[i] + bb, 0.f);
            if (Cf) {
                Cf[m * ldc + n] = v;
            } else {
                const float w = __shfl_xor(v, 1);      // neighbor column n^1
                if (!(lr & 1)) Cp[m * (ldc >> 1) + (n >> 1)] = pkbf(v, w);
            }
        }
    }
}

// ---------------------------------------------------------------------------
// Kernel 2: head MLPs batched over graphs (16 graphs / block).
// ---------------------------------------------------------------------------
__global__ __launch_bounds__(256) void head_kernel(
    const float* __restrict__ hp, HeadP P, float* __restrict__ out)
{
    __shared__ __align__(16) unsigned X0p[GB * 160];  // packed inputs (10KB)
    __shared__ __align__(16) unsigned P1p[GB * 128];  // packed P1 (8KB)
    __shared__ __align__(16) unsigned P2p[GB * 64];
    __shared__ __align__(16) unsigned T1p[GB * 32];
    __shared__ float T2f[GB * 32];
    __shared__ float S1f[GB * 32];
    __shared__ float P3f[GB * 64];
    __shared__ float muL[GB * 8];
    __shared__ float gb1[GB * 64];
    __shared__ float gb2[GB * 32];

    const int tid = threadIdx.x;
    const size_t gbase = (size_t)blockIdx.x * GB;

    // stage inputs, packing f32 pairs -> bf16 dwords
    {
        const float2* src = (const float2*)(hp + gbase * 320);
        for (int i = tid; i < GB * 160; i += 256) {
            const float2 pr = src[i];
            X0p[i] = pkbf(pr.x, pr.y);
        }
    }
    // prefetch the big p1 weight image (global_prefetch) while t/s branches run
    for (int i = tid; i < 640; i += 256)
        __builtin_prefetch(P.p1p + i * 64, 0, 3);
    __syncthreads();

    // topo branch: 128 -> 64 -> 32
    wmma_gemm_pk(X0p, 160, P.t1p, 64, P.t1b, nullptr, T1p, 64, GB / 16, 4, 128);
    __syncthreads();
    wmma_gemm_pk(T1p, 32, P.t2p, 32, P.t2b, T2f, nullptr, 32, GB / 16, 2, 64);
    __syncthreads();
    // struct branch: 192 -> 32 (terminals at column 128 => dword offset 64)
    wmma_gemm_pk(X0p + 64, 160, P.s1p, 96, P.s1b, S1f, nullptr, 32, GB / 16, 2, 192);
    __syncthreads();
    // pz branch: 320 -> 256 -> 128 -> 64
    wmma_gemm_pk(X0p, 160, P.p1p, 160, P.p1b, nullptr, P1p, 256, GB / 16, 16, 320);
    __syncthreads();
    wmma_gemm_pk(P1p, 128, P.p2p, 128, P.p2b, nullptr, P2p, 128, GB / 16, 8, 256);
    __syncthreads();
    wmma_gemm_pk(P2p, 64, P.p3p, 64, P.p3b, P3f, nullptr, 64, GB / 16, 4, 128);
    __syncthreads();

    // mu / logvar assembly (N=2/2/4 heads: VALU)
    float* zout  = out;
    float* muout = out + (size_t)NUM_GRAPHS * 8;
    float* lvout = out + (size_t)NUM_GRAPHS * 16;
    float* gout  = out + (size_t)NUM_GRAPHS * 24;
    if (tid < GB * 8) {
        const int gg = tid >> 3, c = tid & 7;
        float m, v;
        if (c < 2) {
            m = P.tmub[c]; v = P.tlvb[c];
            for (int k = 0; k < 32; ++k) {
                const float h = T2f[gg * 32 + k];
                m += h * P.tmuw[k * 2 + c];
                v += h * P.tlvw[k * 2 + c];
            }
        } else if (c < 4) {
            const int cc = c - 2;
            m = P.smub[cc]; v = P.slvb[cc];
            for (int k = 0; k < 32; ++k) {
                const float h = S1f[gg * 32 + k];
                m += h * P.smuw[k * 2 + cc];
                v += h * P.slvw[k * 2 + cc];
            }
        } else {
            const int cc = c - 4;
            m = P.pmub[cc]; v = P.plvb[cc];
            for (int k = 0; k < 64; ++k) {
                const float h = P3f[gg * 64 + k];
                m += h * P.pmuw[k * 4 + cc];
                v += h * P.plvw[k * 4 + cc];
            }
        }
        muL[tid] = m;
        const size_t go = (gbase + gg) * 8 + c;
        zout[go] = m; muout[go] = m; lvout[go] = v;   // z == mu (eval mode)
    }
    __syncthreads();

    // gain head: 8 -> 64 -> 32 -> 1 (tiny K: VALU)
    for (int i = tid; i < GB * 64; i += 256) {
        const int gg = i >> 6, f = i & 63;
        float s = P.g1b[f];
        for (int k = 0; k < 8; ++k) s += muL[gg * 8 + k] * P.g1w[k * 64 + f];
        gb1[i] = fmaxf(s, 0.f);
    }
    __syncthreads();
    for (int i = tid; i < GB * 32; i += 256) {
        const int gg = i >> 5, f = i & 31;
        float s = P.g2b[f];
        for (int k = 0; k < 64; ++k) s += gb1[gg * 64 + k] * P.g2w[k * 32 + f];
        gb2[i] = fmaxf(s, 0.f);
    }
    __syncthreads();
    if (tid < GB) {
        float s = P.g3b[0];
        for (int k = 0; k < 32; ++k) s += gb2[tid * 32 + k] * P.g3w[k];
        gout[gbase + tid] = s;
    }
}

extern "C" void kernel_launch(void* const* d_in, const int* in_sizes, int n_in,
                              void* d_out, int out_size, void* d_ws, size_t ws_size,
                              hipStream_t stream) {
    (void)in_sizes; (void)n_in; (void)out_size; (void)ws_size;
    const float* x  = (const float*)d_in[0];
    const int*   ei = (const int*)d_in[1];
    const float* ea = (const float*)d_in[2];
    // d_in[3] = batch (implied by fixed layout)
    const float* const* pp = (const float* const*)(d_in + 4);  // params: W,b,A,E x3 then heads
    const float* const* q  = pp + 12;

    float*    wsf = (float*)d_ws;
    unsigned* wsu = (unsigned*)d_ws;

    // ---- pre-pack reused weight matrices into [N][K/2] bf16-pair images ----
    pack_kernel<<<8,   256, 0, stream>>>(pp[4],  wsu + GW1_OFF,  64,  64);  // gnn1_W
    pack_kernel<<<8,   256, 0, stream>>>(pp[6],  wsu + GA1_OFF,  64,  64);  // gnn1_A
    pack_kernel<<<8,   256, 0, stream>>>(pp[8],  wsu + GW2_OFF,  64,  64);  // gnn2_W
    pack_kernel<<<8,   256, 0, stream>>>(pp[10], wsu + GA2_OFF,  64,  64);  // gnn2_A
    pack_kernel<<<16,  256, 0, stream>>>(q[0],   wsu + T1P_OFF, 128,  64);  // t1_w
    pack_kernel<<<4,   256, 0, stream>>>(q[2],   wsu + T2P_OFF,  64,  32);  // t2_w
    pack_kernel<<<12,  256, 0, stream>>>(q[8],   wsu + S1P_OFF, 192,  32);  // s1_w
    pack_kernel<<<160, 256, 0, stream>>>(q[14],  wsu + P1P_OFF, 320, 256);  // p1_w
    pack_kernel<<<64,  256, 0, stream>>>(q[16],  wsu + P2P_OFF, 256, 128);  // p2_w
    pack_kernel<<<16,  256, 0, stream>>>(q[18],  wsu + P3P_OFF, 128,  64);  // p3_w

    GnnP G;
    G.W0 = pp[0]; G.A0 = pp[2];
    for (int l = 0; l < 3; ++l) { G.b[l] = pp[4 * l + 1]; G.E[l] = pp[4 * l + 3]; }
    G.Wp[0] = wsu + GW1_OFF; G.Ap[0] = wsu + GA1_OFF;
    G.Wp[1] = wsu + GW2_OFF; G.Ap[1] = wsu + GA2_OFF;

    HeadP H;
    H.t1p = wsu + T1P_OFF; H.t2p = wsu + T2P_OFF; H.s1p = wsu + S1P_OFF;
    H.p1p = wsu + P1P_OFF; H.p2p = wsu + P2P_OFF; H.p3p = wsu + P3P_OFF;
    H.t1b = q[1];  H.t2b = q[3];  H.s1b = q[9];
    H.p1b = q[15]; H.p2b = q[17]; H.p3b = q[19];
    H.tmuw = q[4];  H.tmub = q[5];  H.tlvw = q[6];  H.tlvb = q[7];
    H.smuw = q[10]; H.smub = q[11]; H.slvw = q[12]; H.slvb = q[13];
    H.pmuw = q[20]; H.pmub = q[21]; H.plvw = q[22]; H.plvb = q[23];
    H.g1w = q[24]; H.g1b = q[25]; H.g2w = q[26]; H.g2b = q[27];
    H.g3w = q[28]; H.g3b = q[29];

    float* hpools = wsf + HP_OFF;   // [NUM_GRAPHS][320]

    gnn_kernel<<<NUM_GRAPHS, 256, 0, stream>>>(x, ei, ea, G, hpools);
    head_kernel<<<NUM_GRAPHS / GB, 256, 0, stream>>>(hpools, H, (float*)d_out);
}
